// TransformerDiffusion_28381143892834
// MI455X (gfx1250) — matrix operand order repaired and checked by reading, not compiled, not run on hardware
//
#include <hip/hip_runtime.h>
#include <hip/hip_bf16.h>
#include <stdint.h>

// Problem constants (match reference)
#define B_     8
#define N_     1024
#define M_     512
#define D_     128
#define HEADS_ 8
#define L_     6
#define DFF_   512
#define S_     (N_ + M_)
#define HD_    (D_ / HEADS_)
#define QLD_   (3 * D_)   // packed QKV row stride

typedef __attribute__((ext_vector_type(16))) _Float16 v16h;
typedef __attribute__((ext_vector_type(8)))  _Float16 v8h;
typedef __attribute__((ext_vector_type(8)))  float    v8f;
typedef __attribute__((ext_vector_type(4)))  float    v4f;

__device__ __forceinline__ v8f wmma_f32_f16(v16h a, v16h b, v8f c) {
    // 8 args: (neg_a, A, neg_b, B, c_mod, C, reuse_a, reuse_b)
    return __builtin_amdgcn_wmma_f32_16x16x32_f16(false, a, false, b, (short)0, c,
                                                  false, false);
}
__device__ __forceinline__ v16h cat8(v8h lo, v8h hi) {
    return __builtin_shufflevector(lo, hi, 0, 1, 2, 3, 4, 5, 6, 7,
                                           8, 9, 10, 11, 12, 13, 14, 15);
}

// ---------------------------------------------------------------------------
// Syndrome: syn[b][m] = xor over n of (bit(r[b,n]) & H[m,n])
// ---------------------------------------------------------------------------
__global__ void syndrome_kernel(const float* __restrict__ r,
                                const int* __restrict__ pcm,
                                float* __restrict__ syn) {
    int idx = blockIdx.x * blockDim.x + threadIdx.x;
    if (idx >= B_ * M_) return;
    int b = idx / M_, m = idx % M_;
    const float* rb = r + (size_t)b * N_;
    const int*   Hm = pcm + (size_t)m * N_;
    int acc = 0;
    for (int n = 0; n < N_; ++n)
        acc ^= (Hm[n] & (rb[n] < 0.f ? 1 : 0));
    syn[idx] = (float)acc;
}

// ---------------------------------------------------------------------------
// x0[b,s,d] = te[b,d] * src_embed[s,d] * node(b,s)
// ---------------------------------------------------------------------------
__global__ __launch_bounds__(128)
void init_x_kernel(const float* __restrict__ r, const float* __restrict__ syn,
                   const float* __restrict__ src_embed,
                   const float* __restrict__ time_table,
                   const int* __restrict__ tt, float* __restrict__ x) {
    int bs = blockIdx.x;
    int b = bs / S_, s = bs % S_;
    int d = threadIdx.x;
    float node = (s < N_) ? fabsf(r[(size_t)b * N_ + s])
                          : syn[(size_t)b * M_ + (s - N_)];
    float te = time_table[(size_t)tt[b] * D_ + d];
    x[(size_t)bs * D_ + d] = te * src_embed[(size_t)s * D_ + d] * node;
}

// ---------------------------------------------------------------------------
// fp32 -> fp16 weight conversion (flat)
// ---------------------------------------------------------------------------
__global__ void f32_to_f16_kernel(const float* __restrict__ in,
                                  _Float16* __restrict__ out, int n) {
    int i = blockIdx.x * blockDim.x + threadIdx.x;
    if (i < n) out[i] = (_Float16)in[i];
}

// Pack Wq|Wk|Wv ([L,D,D] each) into [L, D, 3D] f16 (cols: q | k | v)
__global__ void pack_qkv_w_kernel(const float* __restrict__ wq,
                                  const float* __restrict__ wk,
                                  const float* __restrict__ wv,
                                  _Float16* __restrict__ dst) {
    int i = blockIdx.x * blockDim.x + threadIdx.x;   // over L*D*D
    if (i >= L_ * D_ * D_) return;
    int c = i % D_;
    int r = i / D_;                                  // [0, L*D)
    size_t o = (size_t)r * QLD_;
    dst[o + c]            = (_Float16)wq[i];
    dst[o + D_ + c]       = (_Float16)wk[i];
    dst[o + 2 * D_ + c]   = (_Float16)wv[i];
}

// ---------------------------------------------------------------------------
// LayerNorm, one wave32 per row of D=128 (4 f32 per lane), optional FiLM te.
// ---------------------------------------------------------------------------
template <bool USE_TE>
__global__ __launch_bounds__(256)
void layernorm_kernel(const float* __restrict__ x, const float* __restrict__ g,
                      const float* __restrict__ beta,
                      const float* __restrict__ time_table,
                      const int* __restrict__ tt, _Float16* __restrict__ out) {
    const int lane = threadIdx.x & 31;
    const int wave = threadIdx.x >> 5;
    const int row  = blockIdx.x * 8 + wave;   // [0, B*S)
    const int bidx = row / S_;
    const float* xr = x + (size_t)row * D_;
    v4f xv = *((const v4f*)xr + lane);
    float sum = xv[0] + xv[1] + xv[2] + xv[3];
    float sq  = xv[0]*xv[0] + xv[1]*xv[1] + xv[2]*xv[2] + xv[3]*xv[3];
#pragma unroll
    for (int off = 16; off > 0; off >>= 1) {
        sum += __shfl_xor(sum, off, 32);
        sq  += __shfl_xor(sq,  off, 32);
    }
    float mu  = sum * (1.f / D_);
    float var = sq * (1.f / D_) - mu * mu;
    float rs  = rsqrtf(var + 1e-5f);
    const float* te = time_table + (size_t)tt[bidx] * D_;
#pragma unroll
    for (int j = 0; j < 4; ++j) {
        int d = lane * 4 + j;
        float hv = (xv[j] - mu) * rs * g[d] + beta[d];
        if (USE_TE) hv *= te[d];
        out[(size_t)row * D_ + d] = (_Float16)hv;
    }
}

// ---------------------------------------------------------------------------
// Tiled f16 GEMM, f32 WMMA accumulation.
//   C[Mrows x Ncols] = A[Mrows x K] @ W[K x Ncols]
// Block tile 128x64, 256 threads (8 wave32), each wave 32x32 (2x2 WMMA).
// B tile stored TRANSPOSED in LDS so both A and B fragments are 2x b128 reads.
// MODE 0: store f16   MODE 1: store f16 + ReLU   MODE 2: f32 residual add
// ---------------------------------------------------------------------------
template <int MODE>
__global__ __launch_bounds__(256)
void gemm_kernel(const _Float16* __restrict__ A, const _Float16* __restrict__ W,
                 void* __restrict__ Out, int Mrows, int Ncols, int K) {
    constexpr int SA  = 40;   // padded LDS stride (halfs) for A tile 128x32
    constexpr int SBT = 40;   // padded LDS stride (halfs) for B^T tile 64x32
    __shared__ _Float16 As[128 * SA];
    __shared__ _Float16 BsT[64 * SBT];

    const int tid  = threadIdx.x;
    const int lane = tid & 31;
    const int wave = tid >> 5;
    const int block_row = blockIdx.x * 128;
    const int block_col = blockIdx.y * 64;
    const int wrow = (wave >> 1) * 32;   // 0/32/64/96
    const int wcol = (wave & 1) * 32;    // 0/32

    v8f acc[2][2] = {};

    for (int k0 = 0; k0 < K; k0 += 32) {
        // A tile 128x32: each thread one 32B chunk
        {
            int r = tid >> 1;
            int c = (tid & 1) * 16;
            const _Float16* src = A + (size_t)(block_row + r) * K + k0 + c;
            v8h* dst = (v8h*)(As + r * SA + c);
            dst[0] = ((const v8h*)src)[0];
            dst[1] = ((const v8h*)src)[1];
        }
        // B tile 32x64 -> transposed LDS [col][k]
        {
            int r = tid >> 3;          // K row 0..31
            int c = (tid & 7) * 8;     // col group
            const _Float16* src = W + (size_t)(k0 + r) * Ncols + block_col + c;
            v8h w0 = ((const v8h*)src)[0];
#pragma unroll
            for (int j = 0; j < 8; ++j)
                BsT[(c + j) * SBT + r] = w0[j];
        }
        if (k0 + 32 < K) {   // next K slice -> global_prefetch_b8
            __builtin_prefetch(A + (size_t)(block_row + (tid >> 1)) * K + k0 + 32, 0, 1);
            __builtin_prefetch(W + (size_t)(k0 + 32 + (tid >> 3)) * Ncols + block_col + (tid & 7) * 8, 0, 1);
        }
        __syncthreads();

        // Fragments per CDNA5 16-bit A(16x32)/B(32x16) lane layouts.
        const int kc = (lane >> 4) * 8;    // A: lanes 0-15 hold K0-7 & K16-23
        const int kb = (lane >> 4) * 8;    // B: lanes 0-15 hold K0-7 & K16-23
        v16h afrag[2], bfrag[2];
#pragma unroll
        for (int tm = 0; tm < 2; ++tm) {
            int row = wrow + tm * 16 + (lane & 15);
            v8h alo = *(const v8h*)(As + row * SA + kc);
            v8h ahi = *(const v8h*)(As + row * SA + 16 + kc);
            afrag[tm] = cat8(alo, ahi);
        }
#pragma unroll
        for (int tn = 0; tn < 2; ++tn) {
            int col = wcol + tn * 16 + (lane & 15);
            v8h blo = *(const v8h*)(BsT + col * SBT + kb);
            v8h bhi = *(const v8h*)(BsT + col * SBT + 16 + kb);
            bfrag[tn] = cat8(blo, bhi);
        }
#pragma unroll
        for (int tm = 0; tm < 2; ++tm)
#pragma unroll
            for (int tn = 0; tn < 2; ++tn)
                acc[tm][tn] = wmma_f32_f16(afrag[tm], bfrag[tn], acc[tm][tn]);
        __syncthreads();
    }

    // Epilogue: C layout = lane&15 -> N, (lane>=16 ? +8 : 0)+i -> M
#pragma unroll
    for (int tm = 0; tm < 2; ++tm)
#pragma unroll
        for (int tn = 0; tn < 2; ++tn) {
            int mrow = block_row + wrow + tm * 16 + ((lane >> 4) ? 8 : 0);
            int ncol = block_col + wcol + tn * 16 + (lane & 15);
#pragma unroll
            for (int i = 0; i < 8; ++i) {
                float val = acc[tm][tn][i];
                if (MODE == 0) {
                    ((_Float16*)Out)[(size_t)(mrow + i) * Ncols + ncol] = (_Float16)val;
                } else if (MODE == 1) {
                    ((_Float16*)Out)[(size_t)(mrow + i) * Ncols + ncol] =
                        (_Float16)fmaxf(val, 0.f);
                } else {
                    float* o = (float*)Out + (size_t)(mrow + i) * Ncols + ncol;
                    *o += val;
                }
            }
        }
}

// ---------------------------------------------------------------------------
// Flash attention. One wave per (b, head, 16-row query tile), reading packed
// QKV rows [q | k | v] of stride 384. HD=16 is padded to WMMA K=32 with zero
// fragments (branchless vector selects; all loads unconditional b128s).
// ---------------------------------------------------------------------------
__global__ __launch_bounds__(32)
void attention_kernel(const _Float16* __restrict__ qkv,
                      const uint8_t* __restrict__ mask,
                      _Float16* __restrict__ out) {
    __shared__ float    sbuf[16][17];
    __shared__ _Float16 pbuf[16][24];
    __shared__ _Float16 vbuf[16][24];

    const int lane  = threadIdx.x;
    const int qt    = blockIdx.x;
    const int h     = blockIdx.y;
    const int b     = blockIdx.z;
    const int qbase = qt * 16;
    const int col   = lane & 15;
    const int mb    = (lane >> 4) ? 8 : 0;
    const bool lowhalf = (lane < 16);
    const float inv_sqrt_hd = 0.25f;   // 1/sqrt(16)
    const v8h zv8 = {};

    // Q A-fragment, invariant over key tiles (real K=0..15, pad 16..31 = 0)
    v16h aq;
    {
        int row = qbase + col;
        int kc  = (lane >> 4) * 8;
        const v8h* qp =
            (const v8h*)(qkv + ((size_t)b * S_ + row) * QLD_ + h * HD_ + kc);
        aq = cat8(qp[0], zv8);
    }

    v8f  acc   = {};
    float m_run = -1e30f, l_run = 0.f;

    for (int kt = 0; kt < S_ / 16; ++kt) {
        const int kbase = kt * 16;

        // Stage V tile 16x16 to LDS: one coalesced b128 per lane
        {
            int r = lane >> 1;
            int c = (lane & 1) * 8;
            const v8h* src = (const v8h*)(qkv + ((size_t)b * S_ + kbase + r) * QLD_ +
                                          2 * D_ + h * HD_ + c);
            *(v8h*)(&vbuf[r][c]) = src[0];
        }

        // K^T B-fragment: lane(col)=key, features 8*(lane>>4)+0..7; elements 8-15 = 0.
        v16h bk;
        {
            const v8h* kp = (const v8h*)(qkv + ((size_t)b * S_ + kbase + col) * QLD_ +
                                         D_ + h * HD_ + (lane >> 4) * 8);
            v8h klo = kp[0];
            bk = cat8(klo, zv8);
        }
        v8f zero = {};
        v8f s = wmma_f32_f16(aq, bk, zero);

        // Scale + mask bias, stage scores to LDS (C layout -> row major)
#pragma unroll
        for (int i = 0; i < 8; ++i) {
            int rowg = qbase + mb + i;
            int colg = kbase + col;
            float sv = s[i] * inv_sqrt_hd +
                       (mask[(size_t)rowg * S_ + colg] ? -1e9f : 0.f);
            sbuf[mb + i][col] = sv;
        }
        __syncthreads();

        // Online softmax row stats (row r owned by lane r, r < 16)
        float alpha = 1.f;
        if (lowhalf) {
            float tmax = -1e30f;
#pragma unroll
            for (int j = 0; j < 16; ++j) tmax = fmaxf(tmax, sbuf[lane][j]);
            float m_new = fmaxf(m_run, tmax);
            alpha = __expf(m_run - m_new);
            float rsum = 0.f;
#pragma unroll
            for (int j = 0; j < 16; ++j) {
                float p = __expf(sbuf[lane][j] - m_new);
                rsum += p;
                pbuf[lane][j] = (_Float16)p;
            }
            l_run = l_run * alpha + rsum;
            m_run = m_new;
        }
        __syncthreads();

        // Rescale running output by alpha[row]
#pragma unroll
        for (int i = 0; i < 8; ++i) acc[i] *= __shfl(alpha, mb + i, 32);

        // P A-fragment from LDS (real K = 16 keys, pad to 32)
        v16h ap;
        {
            int kc = (lane >> 4) * 8;
            v8h plo = *(const v8h*)(&pbuf[col][0] + kc);
            ap = cat8(plo, zv8);
        }
        // V B-fragment from staged LDS tile: b[i] = v[key kbase+8*(lane>>4)+i][hd=col], i<8
        v16h bv;
        {
#pragma unroll
            for (int i = 0; i < 8; ++i) bv[i] = vbuf[(lane >> 4) * 8 + i][col];
#pragma unroll
            for (int i = 8; i < 16; ++i) bv[i] = (_Float16)0;
        }
        acc = wmma_f32_f16(ap, bv, acc);
        __syncthreads();   // protect sbuf/pbuf/vbuf reuse
    }

    float inv_l = 1.f;
    if (lowhalf) inv_l = 1.f / l_run;
#pragma unroll
    for (int i = 0; i < 8; ++i) {
        float o = acc[i] * __shfl(inv_l, mb + i, 32);
        out[((size_t)b * S_ + qbase + mb + i) * D_ + h * HD_ + col] = (_Float16)o;
    }
}

// ---------------------------------------------------------------------------
// Final projection: out[row] = dot(x[row,:], fc_w) + fc_b
// ---------------------------------------------------------------------------
__global__ __launch_bounds__(256)
void fc_kernel(const float* __restrict__ x, const float* __restrict__ w,
               const float* __restrict__ bias, float* __restrict__ out) {
    int lane = threadIdx.x & 31, wave = threadIdx.x >> 5;
    int row = blockIdx.x * 8 + wave;
    v4f xv = *((const v4f*)(x + (size_t)row * D_) + lane);
    v4f wv = *((const v4f*)w + lane);
    float s = xv[0]*wv[0] + xv[1]*wv[1] + xv[2]*wv[2] + xv[3]*wv[3];
#pragma unroll
    for (int off = 16; off > 0; off >>= 1) s += __shfl_xor(s, off, 32);
    if (lane == 0) out[row] = s + bias[0];
}

// ---------------------------------------------------------------------------
// Host orchestration
// ---------------------------------------------------------------------------
extern "C" void kernel_launch(void* const* d_in, const int* in_sizes, int n_in,
                              void* d_out, int out_size, void* d_ws, size_t ws_size,
                              hipStream_t stream) {
    (void)in_sizes; (void)n_in; (void)out_size; (void)ws_size;

    const float*   r_t        = (const float*)d_in[0];
    const int*     t          = (const int*)d_in[1];
    const int*     pcm        = (const int*)d_in[2];
    const uint8_t* mask       = (const uint8_t*)d_in[3];
    const float*   src_embed  = (const float*)d_in[4];
    const float*   time_table = (const float*)d_in[5];
    const float*   Wq         = (const float*)d_in[6];
    const float*   Wk         = (const float*)d_in[7];
    const float*   Wv         = (const float*)d_in[8];
    const float*   Wo         = (const float*)d_in[9];
    const float*   W1         = (const float*)d_in[10];
    const float*   W2         = (const float*)d_in[11];
    const float*   g1         = (const float*)d_in[12];
    const float*   b1         = (const float*)d_in[13];
    const float*   g2         = (const float*)d_in[14];
    const float*   b2         = (const float*)d_in[15];
    const float*   fc_w       = (const float*)d_in[16];
    const float*   fc_b       = (const float*)d_in[17];

    const int ROWS = B_ * S_;   // 12288

    char* ws = (char*)d_ws;
    size_t off = 0;
    auto alloc = [&](size_t bytes) -> void* {
        void* p = ws + off;
        off = (off + bytes + 255) & ~(size_t)255;
        return p;
    };
    float*    xf     = (float*)   alloc((size_t)ROWS * D_ * 4);
    _Float16* h16    = (_Float16*)alloc((size_t)ROWS * D_ * 2);
    _Float16* qkv16  = (_Float16*)alloc((size_t)ROWS * QLD_ * 2);
    _Float16* attn16 = (_Float16*)alloc((size_t)ROWS * D_ * 2);
    _Float16* ffn16  = (_Float16*)alloc((size_t)ROWS * DFF_ * 2);
    _Float16* wqkv16 = (_Float16*)alloc((size_t)L_ * D_ * QLD_ * 2);
    _Float16* wo16   = (_Float16*)alloc((size_t)L_ * D_ * D_ * 2);
    _Float16* w116   = (_Float16*)alloc((size_t)L_ * D_ * DFF_ * 2);
    _Float16* w216   = (_Float16*)alloc((size_t)L_ * DFF_ * D_ * 2);
    float*    syn    = (float*)   alloc((size_t)B_ * M_ * 4);

    auto cvt = [&](const float* src, _Float16* dst, int n) {
        f32_to_f16_kernel<<<(n + 255) / 256, 256, 0, stream>>>(src, dst, n);
    };
    pack_qkv_w_kernel<<<(L_ * D_ * D_ + 255) / 256, 256, 0, stream>>>(Wq, Wk, Wv, wqkv16);
    cvt(Wo, wo16, L_ * D_ * D_);
    cvt(W1, w116, L_ * D_ * DFF_);
    cvt(W2, w216, L_ * DFF_ * D_);

    syndrome_kernel<<<(B_ * M_ + 255) / 256, 256, 0, stream>>>(r_t, pcm, syn);
    init_x_kernel<<<B_ * S_, 128, 0, stream>>>(r_t, syn, src_embed, time_table, t, xf);

    dim3 gQKV(ROWS / 128, QLD_ / 64);   // [12288,384]
    dim3 gDD (ROWS / 128, D_ / 64);     // [12288,128]
    dim3 gDF (ROWS / 128, DFF_ / 64);   // [12288,512]

    for (int l = 0; l < L_; ++l) {
        layernorm_kernel<false><<<ROWS / 8, 256, 0, stream>>>(
            xf, g1 + (size_t)l * D_, b1 + (size_t)l * D_, time_table, t, h16);

        gemm_kernel<0><<<gQKV, 256, 0, stream>>>(
            h16, wqkv16 + (size_t)l * D_ * QLD_, qkv16, ROWS, QLD_, D_);

        attention_kernel<<<dim3(S_ / 16, HEADS_, B_), 32, 0, stream>>>(
            qkv16, mask, attn16);

        gemm_kernel<2><<<gDD, 256, 0, stream>>>(
            attn16, wo16 + (size_t)l * D_ * D_, xf, ROWS, D_, D_);

        layernorm_kernel<true><<<ROWS / 8, 256, 0, stream>>>(
            xf, g2 + (size_t)l * D_, b2 + (size_t)l * D_, time_table, t, h16);

        gemm_kernel<1><<<gDF, 256, 0, stream>>>(
            h16, w116 + (size_t)l * D_ * DFF_, ffn16, ROWS, DFF_, D_);
        gemm_kernel<2><<<gDD, 256, 0, stream>>>(
            ffn16, w216 + (size_t)l * DFF_ * D_, xf, ROWS, D_, DFF_);
    }

    fc_kernel<<<ROWS / 8, 256, 0, stream>>>(xf, fc_w, fc_b, (float*)d_out);
}
